// GAT_59115929862612
// MI455X (gfx1250) — matrix operand order repaired and checked by reading, not compiled, not run on hardware
//
#include <hip/hip_runtime.h>
#include <hip/hip_bf16.h>
#include <math.h>

typedef __attribute__((ext_vector_type(16))) _Float16 v16h;
typedef __attribute__((ext_vector_type(8)))  _Float16 v8h;
typedef __attribute__((ext_vector_type(8)))  float    v8f;

#define NNODES 50000
#define NEDGES 800000
#define NFEAT  128
#define NHID   64
#define NHEAD1 4
#define NCLASS 121
#define LD2    128
#define NHEAD2 2
#define K2     256

// ---------------- conversion kernels ----------------

__global__ void cast_x_f16(const float* __restrict__ x, _Float16* __restrict__ xh, long long n) {
    long long i = (long long)blockIdx.x * blockDim.x + threadIdx.x;
    if (i < n) xh[i] = (_Float16)x[i];
}

// W1 [4][128][64] -> W1t [4][64][128]  (Bt: row n holds column n of W, contiguous in K)
__global__ void conv_w1t(const float* __restrict__ W1, _Float16* __restrict__ W1t) {
    int i = blockIdx.x * blockDim.x + threadIdx.x;
    const int per = NHID * NFEAT;
    if (i >= NHEAD1 * per) return;
    int head = i / per, r = i % per;
    int n = r / NFEAT, k = r % NFEAT;
    W1t[i] = (_Float16)W1[(size_t)head * NFEAT * NHID + (size_t)k * NHID + n];
}

// W2 [2][256][121] -> W2t [2][128][256], rows 121..127 zero
__global__ void conv_w2t(const float* __restrict__ W2, _Float16* __restrict__ W2t) {
    int i = blockIdx.x * blockDim.x + threadIdx.x;
    const int per = LD2 * K2;
    if (i >= NHEAD2 * per) return;
    int head = i / per, r = i % per;
    int n = r / K2, k = r % K2;
    float v = (n < NCLASS) ? W2[(size_t)head * K2 * NCLASS + (size_t)k * NCLASS + n] : 0.f;
    W2t[i] = (_Float16)v;
}

// ---------------- WMMA GEMM, N-register-blocked ----------------
// C[head][M][ldc] = A[head?][M][K](f16) x Bt[head][NT*16][K](f16)^T
// One wave owns a 16-row block and ALL NT column tiles: A fragment loaded once
// per k-step, NT back-to-back WMMAs against different B fragments.
template<int NT>
__global__ void wmma_gemm_f16_nb(const _Float16* __restrict__ A, const _Float16* __restrict__ Bt,
                                 float* __restrict__ C, int K, int ldc,
                                 int mtiles, int nheads,
                                 long long strideA, long long strideB, long long strideC) {
    int wave = threadIdx.x >> 5;
    int lane = threadIdx.x & 31;
    long long gw = (long long)blockIdx.x * (blockDim.x >> 5) + wave;
    long long total = (long long)nheads * mtiles;
    if (gw >= total) return;            // uniform per wave: EXEC all-ones inside
    int head = (int)(gw / mtiles);
    int mt   = (int)(gw % mtiles);
    int half = lane >> 4;               // lane group 0-15 / 16-31
    int l16  = lane & 15;

    const _Float16* arow  = A  + (size_t)head * strideA + (size_t)(mt * 16 + l16) * K;
    const _Float16* bbase = Bt + (size_t)head * strideB + (size_t)l16 * K;

    v8f acc[NT];
#pragma unroll
    for (int t = 0; t < NT; ++t) acc[t] = (v8f){};

    for (int k0 = 0; k0 < K; k0 += 32) {
        // A fragment: lanes 0-15 hold K[k0..k0+7] & K[k0+16..k0+23]; lanes 16-31 hold +8
        v8h a0 = *(const v8h*)(arow + k0 + half * 8);
        v8h a1 = *(const v8h*)(arow + k0 + 16 + half * 8);
        v16h a;
#pragma unroll
        for (int i = 0; i < 8; ++i) { a[i] = a0[i]; a[8 + i] = a1[i]; }
#pragma unroll
        for (int t = 0; t < NT; ++t) {
            // B fragment (pre-transposed): lane n reads K[k0+8h..+7] & K[k0+16+8h..+7]
            const _Float16* bp = bbase + (size_t)t * 16 * K + k0 + half * 8;
            v8h b0 = *(const v8h*)(bp);
            v8h b1 = *(const v8h*)(bp + 16);
            v16h b;
#pragma unroll
            for (int i = 0; i < 8; ++i) { b[i] = b0[i]; b[8 + i] = b1[i]; }
            acc[t] = __builtin_amdgcn_wmma_f32_16x16x32_f16(false, a, false, b, (short)0, acc[t], false, false);
        }
    }
    // C layout: VGPR r -> M = mt*16 + r + 8*half, N = t*16 + l16
    float* cbase = C + (size_t)head * strideC + (size_t)(mt * 16 + half * 8) * ldc + l16;
#pragma unroll
    for (int t = 0; t < NT; ++t)
#pragma unroll
        for (int r = 0; r < 8; ++r)
            cbase[(size_t)r * ldc + t * 16] = acc[t][r];
}

// ---------------- attention scores: asrc[n]=h[n].a[:F], adst[n]=h[n].a[F:]
__global__ void attn_scores(const float* __restrict__ h, const float* __restrict__ avec,
                            float* __restrict__ asrc, float* __restrict__ adst,
                            int N, int F, int ld, int twoF) {
    int n = blockIdx.x * blockDim.x + threadIdx.x;
    int head = blockIdx.y;
    if (n >= N) return;
    const float* hr = h + ((size_t)head * N + n) * ld;
    const float* as = avec + (size_t)head * twoF;
    const float* ad = as + F;
    float s = 0.f, d = 0.f;
    for (int f = 0; f < F; ++f) { float v = hr[f]; s += v * as[f]; d += v * ad[f]; }
    asrc[(size_t)head * N + n] = s;
    adst[(size_t)head * N + n] = d;
}

__device__ __forceinline__ unsigned enc_f32(float v) {
    unsigned u = __float_as_uint(v);
    return (u & 0x80000000u) ? ~u : (u | 0x80000000u);
}
__device__ __forceinline__ float dec_f32(unsigned e) {
    return (e & 0x80000000u) ? __uint_as_float(e & 0x7FFFFFFFu) : __uint_as_float(~e);
}

// per-edge leaky-relu score + block-reduced global max per head
__global__ void edge_scores_max(const float* __restrict__ asrc, const float* __restrict__ adst,
                                const long long* __restrict__ src, const long long* __restrict__ dst,
                                float* __restrict__ ebuf, unsigned* __restrict__ emax,
                                int N, long long E) {
    int head = blockIdx.y;
    long long e = (long long)blockIdx.x * blockDim.x + threadIdx.x;
    float v = -1e30f;
    if (e < E) {
        long long s = src[e], d = dst[e];
        float t = asrc[(size_t)head * N + s] + adst[(size_t)head * N + d];
        t = (t > 0.f) ? t : 0.2f * t;                 // leaky_relu, alpha=0.2
        ebuf[(size_t)head * E + e] = t;
        v = t;
    }
    __shared__ unsigned sm[256];
    sm[threadIdx.x] = enc_f32(v);
    __syncthreads();
    for (int s2 = 128; s2 > 0; s2 >>= 1) {
        if ((int)threadIdx.x < s2) {
            unsigned a = sm[threadIdx.x], b = sm[threadIdx.x + s2];
            sm[threadIdx.x] = a > b ? a : b;
        }
        __syncthreads();
    }
    if (threadIdx.x == 0) atomicMax(emax + head, sm[0]);
}

// e = exp(e - max); denom[src] += e
__global__ void edge_exp_denom(float* __restrict__ ebuf, const unsigned* __restrict__ emax,
                               const long long* __restrict__ src, float* __restrict__ denom,
                               int N, long long E) {
    int head = blockIdx.y;
    long long e = (long long)blockIdx.x * blockDim.x + threadIdx.x;
    if (e >= E) return;
    float mx = dec_f32(emax[head]);
    float ee = expf(ebuf[(size_t)head * E + e] - mx);
    ebuf[(size_t)head * E + e] = ee;
    atomicAdd(denom + (size_t)head * N + src[e], ee);
}

// hp[src] += (e/denom[src]) * h[dst], feature-parallel
__global__ void scatter_agg(const float* __restrict__ ebuf, const float* __restrict__ denom,
                            const long long* __restrict__ src, const long long* __restrict__ dst,
                            const float* __restrict__ h, float* __restrict__ hp,
                            int N, long long E, int F, int ld) {
    int head = blockIdx.y;
    long long tid = (long long)blockIdx.x * blockDim.x + threadIdx.x;
    int f = (int)(tid % ld);
    long long e = tid / ld;
    if (e >= E || f >= F) return;
    long long s = src[e], d = dst[e];
    float att = ebuf[(size_t)head * E + e] / denom[(size_t)head * N + s];
    atomicAdd(hp + ((size_t)head * N + s) * ld + f,
              att * h[((size_t)head * N + d) * ld + f]);
}

// h2[n][head*64+f] = f16(elu(hp1[head][n][f] + b1[head][f]))
__global__ void bias_elu_concat(const float* __restrict__ hp1, const float* __restrict__ b1,
                                _Float16* __restrict__ h2, int N) {
    long long i = (long long)blockIdx.x * blockDim.x + threadIdx.x;
    if (i >= (long long)N * 256) return;
    int n = (int)(i / 256), c = (int)(i % 256);
    int head = c >> 6, f = c & 63;
    float v = hp1[((size_t)head * N + n) * NHID + f] + b1[head * NHID + f];
    v = (v > 0.f) ? v : (expf(v) - 1.f);              // elu
    h2[i] = (_Float16)v;
}

// mean over 2 heads + bias, then log_softmax over 121 classes
__global__ void finalize_logsoftmax(const float* __restrict__ hp2, const float* __restrict__ b2,
                                    float* __restrict__ out, int N) {
    int n = blockIdx.x;
    int c = threadIdx.x;           // 128 threads
    float v = -1e30f;
    if (c < NCLASS) {
        float v0 = hp2[((size_t)0 * N + n) * LD2 + c] + b2[c];
        float v1 = hp2[((size_t)1 * N + n) * LD2 + c] + b2[NCLASS + c];
        v = 0.5f * (v0 + v1);
    }
    __shared__ float red[128];
    red[c] = v; __syncthreads();
    for (int s = 64; s > 0; s >>= 1) {
        if (c < s) red[c] = fmaxf(red[c], red[c + s]);
        __syncthreads();
    }
    float mx = red[0]; __syncthreads();
    float ex = (c < NCLASS) ? expf(v - mx) : 0.f;
    red[c] = ex; __syncthreads();
    for (int s = 64; s > 0; s >>= 1) {
        if (c < s) red[c] += red[c + s];
        __syncthreads();
    }
    float lse = logf(red[0]);
    if (c < NCLASS) out[(size_t)n * NCLASS + c] = v - mx - lse;
}

// ---------------- launch ----------------

extern "C" void kernel_launch(void* const* d_in, const int* in_sizes, int n_in,
                              void* d_out, int out_size, void* d_ws, size_t ws_size,
                              hipStream_t stream) {
    const float*      x  = (const float*)d_in[0];
    const long long*  el = (const long long*)d_in[1];   // [2][E] int64
    const float*      W1 = (const float*)d_in[2];
    const float*      a1 = (const float*)d_in[3];
    const float*      b1 = (const float*)d_in[4];
    const float*      W2 = (const float*)d_in[5];
    const float*      a2 = (const float*)d_in[6];
    const float*      b2 = (const float*)d_in[7];
    float* out = (float*)d_out;

    const int N = NNODES;
    const long long E = NEDGES;
    const long long* src = el;
    const long long* dst = el + E;

    // ---- workspace carving (256B aligned) ----
    char* ws = (char*)d_ws;
    size_t off = 0;
    auto carve = [&](size_t bytes) { size_t o = off; off = (off + bytes + 255) & ~(size_t)255; return ws + o; };

    _Float16* xh   = (_Float16*)carve((size_t)N * NFEAT * 2);
    _Float16* w1t  = (_Float16*)carve((size_t)NHEAD1 * NHID * NFEAT * 2);
    float*    h1   = (float*)   carve((size_t)NHEAD1 * N * NHID * 4);   // aliased later as o2 (2*N*128*4 == same size)
    float*    hp1  = (float*)   carve((size_t)NHEAD1 * N * NHID * 4);   // aliased later as hp2
    _Float16* h2   = (_Float16*)carve((size_t)N * 256 * 2);
    _Float16* w2t  = (_Float16*)carve((size_t)NHEAD2 * LD2 * K2 * 2);
    float*    as1  = (float*)   carve((size_t)NHEAD1 * N * 4);
    float*    ad1  = (float*)   carve((size_t)NHEAD1 * N * 4);
    float*    as2  = (float*)   carve((size_t)NHEAD2 * N * 4);
    float*    ad2  = (float*)   carve((size_t)NHEAD2 * N * 4);
    float*    e1   = (float*)   carve((size_t)NHEAD1 * E * 4);
    float*    e2   = (float*)   carve((size_t)NHEAD2 * E * 4);
    float*    den1 = (float*)   carve((size_t)NHEAD1 * N * 4);
    float*    den2 = (float*)   carve((size_t)NHEAD2 * N * 4);
    unsigned* emax = (unsigned*)carve(64);
    float*    o2  = h1;    // reuse: h1 dead after scatter1, o2 written by gemm2 (later in stream)
    float*    hp2 = hp1;   // reuse: hp1 dead after bias_elu, re-zeroed after

    // ---- layer 1 ----
    cast_x_f16<<<(N * NFEAT + 255) / 256, 256, 0, stream>>>(x, xh, (long long)N * NFEAT);
    conv_w1t<<<(NHEAD1 * NHID * NFEAT + 255) / 256, 256, 0, stream>>>(W1, w1t);
    conv_w2t<<<(NHEAD2 * LD2 * K2 + 255) / 256, 256, 0, stream>>>(W2, w2t);

    hipMemsetAsync(hp1,  0, (size_t)NHEAD1 * N * NHID * 4, stream);
    hipMemsetAsync(den1, 0, (size_t)NHEAD1 * N * 4, stream);
    hipMemsetAsync(emax, 0, 64, stream);

    {   // h1 = x @ W1 per head:  M=50000 (3125 tiles), 4 n-tiles per wave, K=128
        long long total = (long long)NHEAD1 * 3125;
        wmma_gemm_f16_nb<4><<<(int)((total + 7) / 8), 256, 0, stream>>>(
            xh, w1t, h1, NFEAT, NHID, 3125, NHEAD1,
            0LL, (long long)NHID * NFEAT, (long long)N * NHID);
    }
    attn_scores<<<dim3((N + 255) / 256, NHEAD1), 256, 0, stream>>>(h1, a1, as1, ad1, N, NHID, NHID, 2 * NHID);
    edge_scores_max<<<dim3((int)((E + 255) / 256), NHEAD1), 256, 0, stream>>>(as1, ad1, src, dst, e1, emax, N, E);
    edge_exp_denom<<<dim3((int)((E + 255) / 256), NHEAD1), 256, 0, stream>>>(e1, emax, src, den1, N, E);
    scatter_agg<<<dim3((int)((E * NHID + 255) / 256), NHEAD1), 256, 0, stream>>>(
        e1, den1, src, dst, h1, hp1, N, E, NHID, NHID);

    // concat + elu -> f16
    bias_elu_concat<<<(int)(((long long)N * 256 + 255) / 256), 256, 0, stream>>>(hp1, b1, h2, N);

    // ---- layer 2 ----
    hipMemsetAsync(hp2,  0, (size_t)NHEAD2 * N * LD2 * 4, stream);
    hipMemsetAsync(den2, 0, (size_t)NHEAD2 * N * 4, stream);
    hipMemsetAsync(emax, 0, 64, stream);  // layer-2 uses slots [4..5]

    {   // o2 = h2 @ W2 per head: M=50000 (3125 tiles), 8 n-tiles per wave (121 pad 128), K=256
        long long total = (long long)NHEAD2 * 3125;
        wmma_gemm_f16_nb<8><<<(int)((total + 7) / 8), 256, 0, stream>>>(
            h2, w2t, o2, K2, LD2, 3125, NHEAD2,
            0LL, (long long)LD2 * K2, (long long)N * LD2);
    }
    attn_scores<<<dim3((N + 255) / 256, NHEAD2), 256, 0, stream>>>(o2, a2, as2, ad2, N, NCLASS, LD2, 2 * NCLASS);
    edge_scores_max<<<dim3((int)((E + 255) / 256), NHEAD2), 256, 0, stream>>>(as2, ad2, src, dst, e2, emax + 4, N, E);
    edge_exp_denom<<<dim3((int)((E + 255) / 256), NHEAD2), 256, 0, stream>>>(e2, emax + 4, src, den2, N, E);
    scatter_agg<<<dim3((int)((E * LD2 + 255) / 256), NHEAD2), 256, 0, stream>>>(
        e2, den2, src, dst, o2, hp2, N, E, NCLASS, LD2);

    finalize_logsoftmax<<<N, 128, 0, stream>>>(hp2, b2, out, N);

    (void)in_sizes; (void)n_in; (void)out_size; (void)ws_size;
}
